// CoefficientDecoder_13649406067079
// MI455X (gfx1250) — matrix operand order repaired and checked by reading, not compiled, not run on hardware
//
#include <hip/hip_runtime.h>
#include <hip/hip_bf16.h>
#include <stddef.h>

// ---------------------------------------------------------------------------
// CDNA5 (gfx1250) fused CoefficientDecoder:
//   z = (((relu(x W1^T + b1)) Wd3^T + bd3) Wd2^T + bd2) Wd1^T + bd1) W2^T + b2
//   out = z @ bases        (the 34-GFLOP GEMM; bf16 WMMA, f32 accumulate,
//                           B tile staged in LDS, A software-pipelined)
// ---------------------------------------------------------------------------

typedef __attribute__((ext_vector_type(16))) __bf16 v16bf;
typedef __attribute__((ext_vector_type(8)))  __bf16 v8bf;
typedef __attribute__((ext_vector_type(8)))  float  v8f;

union AF { v16bf v; v8bf h[2]; };

__device__ __forceinline__ v8f wmma_bf16(AF a, AF b, v8f c) {
    // v_wmma_f32_16x16x32_bf16: (neg_a, A, neg_b, B, c_mod, C, reuse_a, reuse_b)
    return __builtin_amdgcn_wmma_f32_16x16x32_bf16(
        false, a.v, false, b.v, (short)0, c, false, false);
}

// A-fragment (16x32 bf16, MxK) from row-major [M][K] storage, ldk = K stride.
// lane<16: e0..15 = K..K+15 ; lane>=16: e0..15 = K+16..K+31.
__device__ __forceinline__ AF loadA(const __bf16* base, int row, int ldk,
                                    int kstart, int lane) {
    const int m = lane & 15, half = lane >> 4;
    const __bf16* p = base + (size_t)(row + m) * ldk + kstart + half * 16;
    AF a;
    a.h[0] = *(const v8bf*)(p);
    a.h[1] = *(const v8bf*)(p + 8);
    return a;
}

// B-fragment (32x16 bf16, KxN) from [N][K] storage (weights are [out][in]).
// lane = N (mod 16); lanes 0-15 hold K..K+15, lanes 16-31 hold K+16..K+31.
__device__ __forceinline__ AF loadB(const __bf16* baseNK, int ncol, int ldk,
                                    int kstart, int lane) {
    const int n = lane & 15, half = lane >> 4;
    const __bf16* p = baseNK + (size_t)(ncol + n) * ldk + kstart + half * 16;
    AF b;
    b.h[0] = *(const v8bf*)(p);
    b.h[1] = *(const v8bf*)(p + 8);
    return b;
}

// ------------------------------ converters ---------------------------------

__global__ void f32_to_bf16_kernel(const float* __restrict__ in,
                                   __bf16* __restrict__ out, int n) {
    int i = blockIdx.x * blockDim.x + threadIdx.x;
    if (i < n) out[i] = (__bf16)in[i];
}

// bases [512][4096] f32 row-major  ->  basesT [4096][512] bf16
__global__ __launch_bounds__(256)
void transpose_conv_kernel(const float* __restrict__ in,
                           __bf16* __restrict__ outT) {
    __shared__ float tile[32][33];
    const int nx = blockIdx.x * 32;           // N base
    const int ky = blockIdx.y * 32;           // K base
    const int tx = threadIdx.x, ty = threadIdx.y;
    #pragma unroll
    for (int i = 0; i < 32; i += 8)
        tile[ty + i][tx] = in[(size_t)(ky + ty + i) * 4096 + nx + tx];
    __syncthreads();
    #pragma unroll
    for (int i = 0; i < 32; i += 8)
        outT[(size_t)(nx + ty + i) * 512 + ky + tx] = (__bf16)tile[tx][ty + i];
}

// ------------------------------ fused MLP ----------------------------------
// 8 waves/block; each wave owns a 16-row tile. Hidden state (16x64) round-
// trips through LDS to convert C-layout (N across lanes) -> A-fragment layout.

__global__ __launch_bounds__(256)
void mlp_kernel(const __bf16* __restrict__ xbf,
                const __bf16* __restrict__ W1bf, const float* __restrict__ b1,
                const __bf16* __restrict__ Wd3bf, const float* __restrict__ bd3,
                const __bf16* __restrict__ Wd2bf, const float* __restrict__ bd2,
                const __bf16* __restrict__ Wd1bf, const float* __restrict__ bd1,
                const __bf16* __restrict__ W2bf,  const float* __restrict__ b2,
                __bf16* __restrict__ zbf) {
    __shared__ __bf16 hbuf[8 * 16 * 64];      // 16 KB, one 16x64 tile per wave
    const int wv   = threadIdx.x >> 5;
    const int lane = threadIdx.x & 31;
    const int m    = lane & 15;
    const int half = lane >> 4;
    const int mhi  = half * 8;
    const int row0 = blockIdx.x * 128 + wv * 16;
    __bf16* hw = &hbuf[wv * 16 * 64];
    const v8f vzero = {};

    // ---- layer 1: h = relu(x @ W1^T + b1), K=256 ----
    v8f acc[4];
    #pragma unroll
    for (int nt = 0; nt < 4; ++nt) acc[nt] = vzero;
    #pragma unroll
    for (int kb = 0; kb < 8; ++kb) {
        AF a = loadA(xbf, row0, 256, kb * 32, lane);
        #pragma unroll
        for (int nt = 0; nt < 4; ++nt) {
            AF b = loadB(W1bf, nt * 16, 256, kb * 32, lane);
            acc[nt] = wmma_bf16(a, b, acc[nt]);
        }
    }
    #pragma unroll
    for (int nt = 0; nt < 4; ++nt) {
        float bias = b1[nt * 16 + m];
        #pragma unroll
        for (int r = 0; r < 8; ++r) {
            float v = fmaxf(acc[nt][r] + bias, 0.0f);
            hw[(mhi + r) * 64 + nt * 16 + m] = (__bf16)v;
        }
    }

    // ---- three 64x64 dec layers (dec3, dec2, dec1), no activation ----
    const __bf16* Wchain[3] = {Wd3bf, Wd2bf, Wd1bf};
    const float*  bchain[3] = {bd3, bd2, bd1};
    for (int layer = 0; layer < 3; ++layer) {
        AF a0 = loadA(hw, 0, 64, 0,  lane);   // read full A before overwrite
        AF a1 = loadA(hw, 0, 64, 32, lane);
        #pragma unroll
        for (int nt = 0; nt < 4; ++nt) {
            AF b0 = loadB(Wchain[layer], nt * 16, 64, 0,  lane);
            AF b1f = loadB(Wchain[layer], nt * 16, 64, 32, lane);
            acc[nt] = wmma_bf16(a0, b0, vzero);
            acc[nt] = wmma_bf16(a1, b1f, acc[nt]);
        }
        #pragma unroll
        for (int nt = 0; nt < 4; ++nt) {
            float bias = bchain[layer][nt * 16 + m];
            #pragma unroll
            for (int r = 0; r < 8; ++r)
                hw[(mhi + r) * 64 + nt * 16 + m] = (__bf16)(acc[nt][r] + bias);
        }
    }

    // ---- decoder2: z = h @ W2^T + b2, N=512, write bf16 z ----
    AF a0 = loadA(hw, 0, 64, 0,  lane);
    AF a1 = loadA(hw, 0, 64, 32, lane);
    for (int nt = 0; nt < 32; ++nt) {
        AF b0 = loadB(W2bf, nt * 16, 64, 0,  lane);
        AF b1f = loadB(W2bf, nt * 16, 64, 32, lane);
        v8f c = wmma_bf16(a0, b0, vzero);
        c = wmma_bf16(a1, b1f, c);
        float bias = b2[nt * 16 + m];
        #pragma unroll
        for (int r = 0; r < 8; ++r)
            zbf[(size_t)(row0 + mhi + r) * 512 + nt * 16 + m] =
                (__bf16)(c[r] + bias);
    }
}

// ------------------------- big GEMM: out = z @ bases -----------------------
// Block = 8 waves, tile 128 rows x 64 cols. The 64x512 B tile (bases^T, bf16)
// is staged in LDS in two 64x256 halves (32 KB, row-padded to 264 elems so
// ds_load_b128 B-fragment reads are bank-conflict free). A fragments are
// software-pipelined one k-step ahead from global.
#define BP 264   // padded LDS row stride (elements); 264*2B=528B -> 132 dwords

__global__ __launch_bounds__(256)
void gemm_zb_kernel(const __bf16* __restrict__ zbf,
                    const __bf16* __restrict__ basesT,  // [4096][512] bf16
                    float* __restrict__ out) {
    __shared__ __bf16 ldsB[64 * BP];          // ~33 KB
    const int tid  = threadIdx.x;
    const int wv   = tid >> 5;
    const int lane = tid & 31;
    const int m    = lane & 15;
    const int half = lane >> 4;
    const int mhi  = half * 8;
    const int row0 = blockIdx.x * 128 + wv * 16;
    const int col0 = blockIdx.y * 64;
    const v8f vzero = {};

    v8f acc[4];
    #pragma unroll
    for (int nt = 0; nt < 4; ++nt) acc[nt] = vzero;

    for (int hf = 0; hf < 2; ++hf) {
        // ---- cooperative stage of B half-tile: cols [col0,col0+64), K half ----
        // 2048 16-byte chunks, 8 per thread: chunk id -> (col = id/32, kc = id%32)
        #pragma unroll
        for (int i = 0; i < 8; ++i) {
            const int id = tid + i * 256;
            const int c  = id >> 5;
            const int kc = (id & 31) * 8;
            const v8bf chunk = *(const v8bf*)(
                basesT + (size_t)(col0 + c) * 512 + hf * 256 + kc);
            *(v8bf*)(&ldsB[c * BP + kc]) = chunk;
        }
        __syncthreads();

        // ---- 8 WMMA k-steps over this half; A pipelined one step ahead ----
        AF a = loadA(zbf, row0, 512, hf * 256, lane);
        #pragma unroll
        for (int kb = 0; kb < 8; ++kb) {
            AF anext = a;
            if (kb < 7) {
                anext = loadA(zbf, row0, 512, hf * 256 + (kb + 1) * 32, lane);
            } else if (hf == 0) {
                // prefetch first A chunk of the next K-half
                __builtin_prefetch(
                    zbf + (size_t)(row0 + m) * 512 + 256 + half * 8, 0, 0);
            }
            #pragma unroll
            for (int nt = 0; nt < 4; ++nt) {
                const __bf16* bp =
                    &ldsB[(nt * 16 + m) * BP + kb * 32 + half * 16];
                AF b;
                b.h[0] = *(const v8bf*)(bp);
                b.h[1] = *(const v8bf*)(bp + 8);
                acc[nt] = wmma_bf16(a, b, acc[nt]);
            }
            a = anext;
        }
        __syncthreads();   // protect LDS before next half overwrites it
    }

    #pragma unroll
    for (int nt = 0; nt < 4; ++nt)
        #pragma unroll
        for (int r = 0; r < 8; ++r)
            out[(size_t)(row0 + mhi + r) * 4096 + col0 + nt * 16 + m] =
                acc[nt][r];
}

// ------------------------------ launcher -----------------------------------

extern "C" void kernel_launch(void* const* d_in, const int* in_sizes, int n_in,
                              void* d_out, int out_size, void* d_ws,
                              size_t ws_size, hipStream_t stream) {
    (void)in_sizes; (void)n_in; (void)out_size; (void)ws_size;
    const float* x     = (const float*)d_in[0];
    const float* W1    = (const float*)d_in[1];
    const float* b1    = (const float*)d_in[2];
    const float* Wd1   = (const float*)d_in[3];
    const float* bd1   = (const float*)d_in[4];
    const float* Wd2   = (const float*)d_in[5];
    const float* bd2   = (const float*)d_in[6];
    const float* Wd3   = (const float*)d_in[7];
    const float* bd3   = (const float*)d_in[8];
    const float* W2    = (const float*)d_in[9];
    const float* b2    = (const float*)d_in[10];
    const float* bases = (const float*)d_in[11];
    float* out = (float*)d_out;

    // workspace layout (bytes)
    char* ws = (char*)d_ws;
    constexpr size_t OFF_XBF = 0;                        // 8192*256*2
    constexpr size_t OFF_W1  = OFF_XBF + 8192ull*256*2;  // 64*256*2
    constexpr size_t OFF_WD1 = OFF_W1  + 64ull*256*2;    // 64*64*2
    constexpr size_t OFF_WD2 = OFF_WD1 + 64ull*64*2;
    constexpr size_t OFF_WD3 = OFF_WD2 + 64ull*64*2;
    constexpr size_t OFF_W2  = OFF_WD3 + 64ull*64*2;     // 512*64*2
    constexpr size_t OFF_Z   = OFF_W2  + 512ull*64*2;    // 8192*512*2
    constexpr size_t OFF_BT  = OFF_Z   + 8192ull*512*2;  // 4096*512*2

    __bf16* xbf   = (__bf16*)(ws + OFF_XBF);
    __bf16* W1bf  = (__bf16*)(ws + OFF_W1);
    __bf16* Wd1bf = (__bf16*)(ws + OFF_WD1);
    __bf16* Wd2bf = (__bf16*)(ws + OFF_WD2);
    __bf16* Wd3bf = (__bf16*)(ws + OFF_WD3);
    __bf16* W2bf  = (__bf16*)(ws + OFF_W2);
    __bf16* zbf   = (__bf16*)(ws + OFF_Z);
    __bf16* bT    = (__bf16*)(ws + OFF_BT);

    auto conv = [&](const float* in, __bf16* o, int n) {
        f32_to_bf16_kernel<<<(n + 255) / 256, 256, 0, stream>>>(in, o, n);
    };
    conv(x,   xbf,   8192 * 256);
    conv(W1,  W1bf,  64 * 256);
    conv(Wd1, Wd1bf, 64 * 64);
    conv(Wd2, Wd2bf, 64 * 64);
    conv(Wd3, Wd3bf, 64 * 64);
    conv(W2,  W2bf,  512 * 64);

    transpose_conv_kernel<<<dim3(4096 / 32, 512 / 32), dim3(32, 8), 0, stream>>>(
        bases, bT);

    mlp_kernel<<<8192 / 128, 256, 0, stream>>>(
        xbf, W1bf, b1, Wd3bf, bd3, Wd2bf, bd2, Wd1bf, bd1, W2bf, b2, zbf);

    gemm_zb_kernel<<<dim3(8192 / 128, 4096 / 64), 256, 0, stream>>>(
        zbf, bT, out);
}
